// quantization_layer_63556926046439
// MI455X (gfx1250) — matrix operand order repaired and checked by reading, not compiled, not run on hardware
//
#include <hip/hip_runtime.h>

typedef __attribute__((ext_vector_type(16))) _Float16 v16h;
typedef __attribute__((ext_vector_type(8)))  float    v8f;

#define D_DIM 512
#define ROWS_PER_BLOCK 128       // 8 waves * 16 rows
#define KCHUNK 32                // codes resident per LDS buffer
#define CHUNK_BYTES (KCHUNK * D_DIM * 2)   // 32 KB of f16

// ---------------------------------------------------------------------------
// Kernel 1: per-code squared norms + f16 codebook copy (hoists all f32->f16
// conversion out of the GEMM hot loop; f16 codebook = 1 MB, L2-resident).
// grid = K blocks, 128 threads; each thread handles 4 floats.
// ---------------------------------------------------------------------------
__global__ void prep_kernel(const float* __restrict__ embed,
                            float* __restrict__ wnorm,
                            _Float16* __restrict__ embF16) {
    __shared__ float sred[4];
    const int k = blockIdx.x;
    const int c = threadIdx.x * 4;
    const float4 v = *(const float4*)(embed + (size_t)k * D_DIM + c);
    _Float16* dst = embF16 + (size_t)k * D_DIM + c;
    dst[0] = (_Float16)v.x;
    dst[1] = (_Float16)v.y;
    dst[2] = (_Float16)v.z;
    dst[3] = (_Float16)v.w;
    float s = v.x * v.x + v.y * v.y + v.z * v.z + v.w * v.w;
    #pragma unroll
    for (int m = 16; m >= 1; m >>= 1) s += __shfl_xor(s, m, 32);
    if ((threadIdx.x & 31) == 0) sred[threadIdx.x >> 5] = s;
    __syncthreads();
    if (threadIdx.x == 0)
        wnorm[k] = sred[0] + sred[1] + sred[2] + sred[3];
}

// ---------------------------------------------------------------------------
// Kernel 2: WMMA distance GEMM + argmin + gather + partial loss.
// Double-buffered LDS codebook chunks filled with GLOBAL_LOAD_ASYNC_TO_LDS.
// Inner loop interleaves both 16-code tiles (2 independent WMMA chains) with
// explicit depth-1 B prefetch so each ds_load has ~2 WMMAs + 1 iteration of
// slack before its s_wait_dscnt.
// grid = N/128 blocks, 256 threads (8 waves, each wave owns 16 rows).
// ---------------------------------------------------------------------------
__global__ void __launch_bounds__(256, 4)
vq_kernel(const float* __restrict__ x,
          const float* __restrict__ embed,
          const float* __restrict__ wnorm,
          const _Float16* __restrict__ embF16,
          float* __restrict__ out,
          float* __restrict__ partials,
          int kChunks) {
    __shared__ _Float16 ldsE[2][KCHUNK * D_DIM];   // 2 x 32 KB (full 64 KB)
    // Post-loop scratch aliased into buffer 0 (last chunk computes from buf 1,
    // and these are only touched after the final barrier of the main loop).
    int*   idxLds = (int*)&ldsE[0][0];                      // 128 ints
    float* sred   = (float*)((char*)&ldsE[0][0] + 512);     // 8 floats

    const int tid  = threadIdx.x;
    const int wid  = tid >> 5;
    const int lane = tid & 31;
    const int m    = lane & 15;   // row (A) / code (B) within tile
    const int h    = lane >> 4;   // half-wave
    const int blockRowBase = blockIdx.x * ROWS_PER_BLOCK;
    const int rowBase      = blockRowBase + wid * 16;

    // ---- Load A tile (16 x 512 f32 -> f16 WMMA fragments, held in VGPRs) ----
    // A fragment, contiguous k order used here: lane m=l%16, a[j] = A[m][16h + j] (j<8),
    //                        a[8+j] = A[m][16h + 8 + j].
    v16h afrag[16];
    {
        const float* xrow = x + (size_t)(rowBase + m) * D_DIM;
        #pragma unroll
        for (int t = 0; t < 16; ++t) {
            const float* p0 = xrow + 32 * t + 16 * h;
            const float* p1 = p0 + 8;
            v16h a;
            #pragma unroll
            for (int j = 0; j < 8; ++j) {
                a[j]     = (_Float16)p0[j];
                a[8 + j] = (_Float16)p1[j];
            }
            afrag[t] = a;
        }
    }

    // Per-lane running argmin over codes with (code % 16) == m
    float bestd[8];
    int   besti[8];
    #pragma unroll
    for (int r = 0; r < 8; ++r) { bestd[r] = 3.4e38f; besti[r] = 0; }

    // LDS byte offsets for async fills (flat-aperture low 32 bits == LDS addr)
    const unsigned thrByte  = (unsigned)tid * 16u;          // 256 thr * 16 B
    const unsigned ldsBase0 = (unsigned)(uintptr_t)&ldsE[0][0];
    const unsigned ldsBase1 = (unsigned)(uintptr_t)&ldsE[1][0];

    // ---- Prologue: async-fill buffer 0 with chunk 0 ----
    {
        const unsigned lo = ldsBase0 + thrByte;
        const unsigned go = thrByte;
        #pragma unroll
        for (int j = 0; j < 8; ++j) {
            asm volatile("global_load_async_to_lds_b128 %0, %1, %2"
                         :: "v"(lo + j * 4096u), "v"(go + j * 4096u),
                            "s"(embF16)
                         : "memory");
        }
        asm volatile("s_wait_asynccnt 0x0" ::: "memory");
    }
    __syncthreads();

    // ---- Sweep codebook: compute on buf (kc&1), async-prefetch next chunk ----
    for (int kc = 0; kc < kChunks; ++kc) {
        const _Float16* bufp = ldsE[kc & 1];

        if (kc + 1 < kChunks) {  // prefetch chunk kc+1 into the other buffer
            const unsigned lo = ((kc & 1) ? ldsBase0 : ldsBase1) + thrByte;
            const unsigned go = (unsigned)(kc + 1) * CHUNK_BYTES + thrByte;
            #pragma unroll
            for (int j = 0; j < 8; ++j) {
                asm volatile("global_load_async_to_lds_b128 %0, %1, %2"
                             :: "v"(lo + j * 4096u), "v"(go + j * 4096u),
                                "s"(embF16)
                             : "memory");
            }
        }

        // B fragment, same contiguous k order as A: lane n=l%16, b[e] = B[16h + e][n]
        //              -> contiguous 16 halfs of code row at 32t + 16h.
        // Two code tiles (nt=0: codes kcBase+m, nt=1: codes kcBase+16+m)
        // interleaved as two independent WMMA chains with depth-1 B prefetch.
        const int   kcBase = kc * KCHUNK;
        const float wn0 = wnorm[kcBase + m];
        const float wn1 = wnorm[kcBase + 16 + m];
        const _Float16* p0 = &bufp[(m)      * D_DIM + 16 * h];
        const _Float16* p1 = &bufp[(16 + m) * D_DIM + 16 * h];

        v8f acc0 = {}, acc1 = {};
        v16h b0 = *(const v16h*)(p0);
        v16h b1 = *(const v16h*)(p1);
        #pragma unroll
        for (int t = 0; t < 16; ++t) {
            v16h n0, n1;
            if (t + 1 < 16) {
                n0 = *(const v16h*)(p0 + 32 * (t + 1));
                n1 = *(const v16h*)(p1 + 32 * (t + 1));
            }
            acc0 = __builtin_amdgcn_wmma_f32_16x16x32_f16(
                false, afrag[t], false, b0, (short)0, acc0, false, false);
            acc1 = __builtin_amdgcn_wmma_f32_16x16x32_f16(
                false, afrag[t], false, b1, (short)0, acc1, false, false);
            b0 = n0; b1 = n1;
        }

        #pragma unroll
        for (int r = 0; r < 8; ++r) {           // row M = r + 8h
            const float c0 = wn0 - 2.0f * acc0[r];
            if (c0 < bestd[r]) { bestd[r] = c0; besti[r] = kcBase + m; }
            const float c1 = wn1 - 2.0f * acc1[r];
            if (c1 < bestd[r]) { bestd[r] = c1; besti[r] = kcBase + 16 + m; }
        }

        asm volatile("s_wait_asynccnt 0x0" ::: "memory");
        __syncthreads();
    }

    // ---- Cross-lane argmin within each 16-lane half (tie -> lower index) ----
    #pragma unroll
    for (int r = 0; r < 8; ++r) {
        #pragma unroll
        for (int mk = 1; mk < 16; mk <<= 1) {
            const float od = __shfl_xor(bestd[r], mk, 32);
            const int   oi = __shfl_xor(besti[r], mk, 32);
            if (od < bestd[r] || (od == bestd[r] && oi < besti[r])) {
                bestd[r] = od; besti[r] = oi;
            }
        }
        if (m == 0) idxLds[wid * 16 + r + 8 * h] = besti[r];
    }
    __syncthreads();

    // ---- Phase 3: f32 row gather (== straight-through output) + loss ----
    float sumsq = 0.0f;
    for (int i = tid; i < ROWS_PER_BLOCK * (D_DIM / 4); i += 256) {
        const int r   = i / (D_DIM / 4);
        const int c   = (i % (D_DIM / 4)) * 4;
        const int row = blockRowBase + r;
        const int code = idxLds[r];
        const float4 e  = *(const float4*)(embed + (size_t)code * D_DIM + c);
        const float4 xv = *(const float4*)(x + (size_t)row * D_DIM + c);
        *(float4*)(out + (size_t)row * D_DIM + c) = e;
        const float d0 = xv.x - e.x, d1 = xv.y - e.y;
        const float d2 = xv.z - e.z, d3 = xv.w - e.w;
        sumsq += d0 * d0 + d1 * d1 + d2 * d2 + d3 * d3;
    }
    #pragma unroll
    for (int mk = 16; mk >= 1; mk >>= 1) sumsq += __shfl_xor(sumsq, mk, 32);
    if (lane == 0) sred[wid] = sumsq;
    __syncthreads();
    if (tid == 0) {
        float t = 0.0f;
        #pragma unroll
        for (int i = 0; i < 8; ++i) t += sred[i];
        partials[blockIdx.x] = t;
    }
}

// ---------------------------------------------------------------------------
// Kernel 3: deterministic reduction of per-block partials -> loss scalar
// ---------------------------------------------------------------------------
__global__ void finalize_kernel(const float* __restrict__ partials, int n,
                                float* __restrict__ lossOut, float scale) {
    __shared__ float sred[8];
    float s = (threadIdx.x < n) ? partials[threadIdx.x] : 0.0f;
    #pragma unroll
    for (int mk = 16; mk >= 1; mk >>= 1) s += __shfl_xor(s, mk, 32);
    if ((threadIdx.x & 31) == 0) sred[threadIdx.x >> 5] = s;
    __syncthreads();
    if (threadIdx.x == 0) {
        float t = 0.0f;
        #pragma unroll
        for (int i = 0; i < 8; ++i) t += sred[i];
        *lossOut = t * scale;
    }
}

extern "C" void kernel_launch(void* const* d_in, const int* in_sizes, int n_in,
                              void* d_out, int out_size, void* d_ws, size_t ws_size,
                              hipStream_t stream) {
    (void)n_in; (void)out_size; (void)ws_size;
    const float* x     = (const float*)d_in[0];
    const float* embed = (const float*)d_in[1];
    float* out = (float*)d_out;

    const int N = in_sizes[0] / D_DIM;   // 32768
    const int K = in_sizes[1] / D_DIM;   // 1024
    const int blocks = N / ROWS_PER_BLOCK;

    // workspace layout: [wnorm: K floats][partials: blocks floats][embF16: K*D halfs]
    float*    wnorm    = (float*)d_ws;
    float*    partials = wnorm + K;
    _Float16* embF16   = (_Float16*)(partials + blocks);

    prep_kernel<<<K, 128, 0, stream>>>(embed, wnorm, embF16);
    vq_kernel<<<blocks, 256, 0, stream>>>(x, embed, wnorm, embF16, out,
                                          partials, K / KCHUNK);
    // loss = 1.25 * sum((x - emb)^2) / (N*D)
    const float scale = 1.25f / ((float)N * (float)D_DIM);
    finalize_kernel<<<1, 256, 0, stream>>>(partials, blocks,
                                           out + (size_t)N * D_DIM, scale);
}
